// MultiheadCrossAttention_78176994722120
// MI455X (gfx1250) — matrix operand order repaired and checked by reading, not compiled, not run on hardware
//
#include <hip/hip_runtime.h>
#include <hip/hip_bf16.h>
#include <hip/hip_fp16.h>

typedef __attribute__((ext_vector_type(16))) _Float16 v16h;
typedef __attribute__((ext_vector_type(8)))  _Float16 v8h;
typedef __attribute__((ext_vector_type(4)))  _Float16 v4h;
typedef __attribute__((ext_vector_type(8)))  float    v8f;
typedef __attribute__((ext_vector_type(4)))  float    v4f;

#define WMMA16(A, B, C) \
  __builtin_amdgcn_wmma_f32_16x16x32_f16(false, (A), false, (B), (short)0, (C), false, false)

constexpr int Bdim = 2, Tdim = 2048, Sdim = 2048, Wdim = 1024, Hn = 16, Dh = 64;
constexpr float QK_SCALE = 0.35355339059327373f; // 64^-0.25 (applied to q and to k)

static __device__ inline v8f vzero8() {
  v8f z;
#pragma unroll
  for (int e = 0; e < 8; ++e) z[e] = 0.0f;
  return z;
}

static __device__ inline v16h cat8(v8h lo, v8h hi) {
  return __builtin_shufflevector(lo, hi, 0, 1, 2, 3, 4, 5, 6, 7,
                                 8, 9, 10, 11, 12, 13, 14, 15);
}

// CDNA5 async global->LDS copy (16B per lane), tracked by ASYNCcnt.
// INST_OFFSET is added to both the LDS and the global address (ISA 15.18.3).
static __device__ inline void async_b128(unsigned lds_addr, const void* gaddr) {
  asm volatile("global_load_async_to_lds_b128 %0, %1, off"
               :: "v"(lds_addr), "v"(gaddr) : "memory");
}
static __device__ inline void async_b128_off16(unsigned lds_addr, const void* gaddr) {
  asm volatile("global_load_async_to_lds_b128 %0, %1, off offset:16"
               :: "v"(lds_addr), "v"(gaddr) : "memory");
}

// ---------------------------------------------------------------------------
// f32 -> f16 elementwise convert with scale (vec4), used for Kh and Wh
// ---------------------------------------------------------------------------
__global__ __launch_bounds__(256) void cvt_f16_scaled(const float* __restrict__ src,
                                                      _Float16* __restrict__ dst,
                                                      float scale, int n4) {
  int i = blockIdx.x * blockDim.x + threadIdx.x;
  if (i < n4) {
    v4f a = reinterpret_cast<const v4f*>(src)[i];
    v4h h;
#pragma unroll
    for (int e = 0; e < 4; ++e) h[e] = (_Float16)(a[e] * scale);
    reinterpret_cast<v4h*>(dst)[i] = h;
  }
}

// ---------------------------------------------------------------------------
// V [b, s, h, d] f32  ->  Vt [b, h, d, s] f16   (64x64 LDS tile transpose)
// grid: (S/64, H, B), 256 threads
// ---------------------------------------------------------------------------
__global__ __launch_bounds__(256) void transpose_v(const float* __restrict__ v,
                                                   _Float16* __restrict__ vt) {
  __shared__ _Float16 tile[64 * 72];
  const int b = blockIdx.z, h = blockIdx.y, sb = blockIdx.x;
  const int t = threadIdx.x;
  const int r = t >> 2, seg = t & 3;

  // load phase: r = s-row within tile, seg picks 16 d-columns
  const float* src = v + ((size_t)(b * Sdim + sb * 64 + r) * Wdim + h * Dh + seg * 16);
#pragma unroll
  for (int q = 0; q < 2; ++q) {
    v4f a = reinterpret_cast<const v4f*>(src)[2 * q];
    v4f c = reinterpret_cast<const v4f*>(src)[2 * q + 1];
    v8h hh;
#pragma unroll
    for (int e = 0; e < 4; ++e) { hh[e] = (_Float16)a[e]; hh[4 + e] = (_Float16)c[e]; }
    *reinterpret_cast<v8h*>(&tile[r * 72 + seg * 16 + q * 8]) = hh;
  }
  __syncthreads();

  // write phase: r = d-row, seg picks 16 s-columns; gather column of tile
  _Float16 vals[16];
#pragma unroll
  for (int e = 0; e < 16; ++e) vals[e] = tile[(seg * 16 + e) * 72 + r];
  _Float16* dst = vt + ((size_t)((b * Hn + h) * Dh + r) * Sdim + sb * 64 + seg * 16);
  v8h w0, w1;
#pragma unroll
  for (int e = 0; e < 8; ++e) { w0[e] = vals[e]; w1[e] = vals[8 + e]; }
  reinterpret_cast<v8h*>(dst)[0] = w0;
  reinterpret_cast<v8h*>(dst)[1] = w1;
}

// ---------------------------------------------------------------------------
// Flash attention.  grid: (T/128, H, B), 256 threads (8 waves x 16 q-rows).
// Computes St = K.Q^T so softmaxed P lands directly in WMMA A-layout.
// K/V chunks are double-buffered in LDS and filled with async copies.
// ---------------------------------------------------------------------------
__global__ __launch_bounds__(256) void flash_attn(const float* __restrict__ x,
                                                  const _Float16* __restrict__ Kh,
                                                  const _Float16* __restrict__ Vt,
                                                  _Float16* __restrict__ Att) {
  __shared__ _Float16 Kl[2][64 * 72];  // [buf][s within chunk][d], padded
  __shared__ _Float16 Vl[2][64 * 72];  // [buf][d][s within chunk], padded
  constexpr unsigned BUFB = 64 * 72 * sizeof(_Float16);  // 9216 bytes

  const int b = blockIdx.z, h = blockIdx.y, mb = blockIdx.x;
  const int t = threadIdx.x;
  const int w = t >> 5, lane = t & 31;
  const int half = lane >> 4, lm = lane & 15;
  const int m0 = mb * 128 + w * 16;

  // ---- Q B-fragments (B-layout: lane = m, regs = K(d) = half*16 + 0..15) ----
  v16h qf[2];
  {
    const float* qp = x + ((size_t)(b * Tdim + m0 + lm) * Wdim + h * Dh + half * 16);
#pragma unroll
    for (int ks = 0; ks < 2; ++ks) {
      const float* p = qp + ks * 32;
      v16h f;
#pragma unroll
      for (int q = 0; q < 4; ++q) {
        v4f a = reinterpret_cast<const v4f*>(p)[q];
#pragma unroll
        for (int e = 0; e < 4; ++e) f[q * 4 + e] = (_Float16)(a[e] * QK_SCALE);
      }
      qf[ks] = f;
    }
  }

  v8f acc[4];
#pragma unroll
  for (int dt = 0; dt < 4; ++dt) acc[dt] = vzero8();
  float mrow = -1e30f, lrow = 0.0f;

  // staging role: r = row (s for K, d for Vt), seg = 16-elem segment (32B)
  const int r = t >> 2, seg = t & 3;
  const _Float16* kg = Kh + ((size_t)(b * Sdim + r) * Wdim + h * Dh + seg * 16);
  const _Float16* vg = Vt + ((size_t)((b * Hn + h) * Dh + r) * Sdim + seg * 16);
  const unsigned kLds0 = (unsigned)(size_t)&Kl[0][r * 72 + seg * 16];
  const unsigned vLds0 = (unsigned)(size_t)&Vl[0][r * 72 + seg * 16];

  // prefetch chunk 0 into buffer 0 (4 async b128 per thread per chunk)
  {
    const void* kp = (const void*)(kg);
    const void* vp = (const void*)(vg);
    async_b128(kLds0, kp);
    async_b128_off16(kLds0, kp);
    async_b128(vLds0, vp);
    async_b128_off16(vLds0, vp);
  }

  int buf = 0;
  for (int sc = 0; sc < Sdim; sc += 64) {
    if (sc + 64 < Sdim) {
      // prefetch next chunk into the other buffer (safe: prior tail barrier
      // guarantees every wave is done reading it)
      const unsigned nb = (unsigned)(buf ^ 1) * BUFB;
      const void* kp = (const void*)(kg + (size_t)(sc + 64) * Wdim);
      const void* vp = (const void*)(vg + (sc + 64));
      async_b128(kLds0 + nb, kp);
      async_b128_off16(kLds0 + nb, kp);
      async_b128(vLds0 + nb, vp);
      async_b128_off16(vLds0 + nb, vp);
      // async loads retire in order: <=4 outstanding means chunk sc landed
      asm volatile("s_wait_asynccnt 4" ::: "memory");
    } else {
      asm volatile("s_wait_asynccnt 0" ::: "memory");
    }
    __syncthreads();

    const _Float16* Kb = Kl[buf];
    const _Float16* Vb = Vl[buf];

    // ---- scores: St[s,m] = sum_d K[s,d]*Q[m,d]  (4 s-tiles of 16) ----
    v8f st[4];
#pragma unroll
    for (int stile = 0; stile < 4; ++stile) {
      v8f c = vzero8();
#pragma unroll
      for (int ks = 0; ks < 2; ++ks) {
        // A-layout: lane = s-row, K(d) = half*16 + 0..15
        const _Float16* kp = &Kb[(stile * 16 + lm) * 72 + ks * 32 + half * 16];
        v16h af = cat8(*reinterpret_cast<const v8h*>(kp),
                       *reinterpret_cast<const v8h*>(kp + 8));
        c = WMMA16(af, qf[ks], c);
      }
      st[stile] = c;
    }

    // ---- online softmax (per q-column lm; halves combined via shfl_xor 16) ----
    float cm = -1e30f;
#pragma unroll
    for (int stile = 0; stile < 4; ++stile)
#pragma unroll
      for (int e = 0; e < 8; ++e) cm = fmaxf(cm, st[stile][e]);
    cm = fmaxf(cm, __shfl_xor(cm, 16));
    const float mnew = fmaxf(mrow, cm);
    const float alpha = __expf(mrow - mnew);
    mrow = mnew;

    float ls = 0.0f;
    v16h pf[2];  // P in A-layout: pf[ks] = pack(st[2ks]) ++ pack(st[2ks+1])
#pragma unroll
    for (int stile = 0; stile < 4; ++stile)
#pragma unroll
      for (int e = 0; e < 8; ++e) {
        float p = __expf(st[stile][e] - mnew);
        ls += p;
        pf[stile >> 1][(stile & 1) * 8 + e] = (_Float16)p;
      }
    ls += __shfl_xor(ls, 16);
    lrow = lrow * alpha + ls;

    // broadcast alpha to accumulator-row layout (row = e + 8*half)
    v8f av;
#pragma unroll
    for (int e = 0; e < 8; ++e) av[e] = __shfl(alpha, e + half * 8);
#pragma unroll
    for (int dt = 0; dt < 4; ++dt) acc[dt] *= av;

    // ---- out += P.V  (B-frag: lane = d-col, regs = K(s) {0..7,16..23} / {8..15,24..31}) ----
#pragma unroll
    for (int ks = 0; ks < 2; ++ks)
#pragma unroll
      for (int dt = 0; dt < 4; ++dt) {
        const _Float16* vp = &Vb[(dt * 16 + lm) * 72 + ks * 32 + half * 8];
        v16h bf = cat8(*reinterpret_cast<const v8h*>(vp),
                       *reinterpret_cast<const v8h*>(vp + 16));
        acc[dt] = WMMA16(pf[ks], bf, acc[dt]);
      }
    __syncthreads();
    buf ^= 1;
  }

  // ---- normalize and store f16 attention output ----
  v8f lv;
#pragma unroll
  for (int e = 0; e < 8; ++e) lv[e] = __shfl(lrow, e + half * 8);
#pragma unroll
  for (int dt = 0; dt < 4; ++dt) {
    const int col = h * Dh + dt * 16 + lm;
#pragma unroll
    for (int e = 0; e < 8; ++e) {
      const int row = m0 + e + half * 8;
      Att[(size_t)(b * Tdim + row) * Wdim + col] = (_Float16)(acc[dt][e] / lv[e]);
    }
  }
}

// ---------------------------------------------------------------------------
// out[M,W] = Att[M,W] @ w_proj^T + b.  grid: (M/64, W/128), 256 threads.
// Wave tile 16x64; w_proj row-major is exactly the K-major B operand.
// ---------------------------------------------------------------------------
__global__ __launch_bounds__(256) void proj_gemm(const _Float16* __restrict__ A,
                                                 const _Float16* __restrict__ Wh,
                                                 const float* __restrict__ bias,
                                                 float* __restrict__ out) {
  const int t = threadIdx.x;
  const int w = t >> 5, lane = t & 31;
  const int half = lane >> 4, lm = lane & 15;
  const int m0 = blockIdx.x * 64 + (w & 3) * 16;
  const int n0 = blockIdx.y * 128 + (w >> 2) * 64;

  v8f acc[4];
#pragma unroll
  for (int nt = 0; nt < 4; ++nt) acc[nt] = vzero8();

  const _Float16* ap = A + (size_t)(m0 + lm) * Wdim + half * 16;
  for (int kc = 0; kc < Wdim; kc += 32) {
    v16h af = cat8(*reinterpret_cast<const v8h*>(ap + kc),
                   *reinterpret_cast<const v8h*>(ap + kc + 8));
#pragma unroll
    for (int nt = 0; nt < 4; ++nt) {
      const _Float16* bp = Wh + (size_t)(n0 + nt * 16 + lm) * Wdim + kc + half * 16;
      v16h bf = cat8(reinterpret_cast<const v8h*>(bp)[0],
                     reinterpret_cast<const v8h*>(bp)[1]);
      acc[nt] = WMMA16(af, bf, acc[nt]);
    }
  }

#pragma unroll
  for (int nt = 0; nt < 4; ++nt) {
    const int n = n0 + nt * 16 + lm;
    const float bb = bias[n];
#pragma unroll
    for (int e = 0; e < 8; ++e) {
      const int row = m0 + e + half * 8;
      out[(size_t)row * Wdim + n] = acc[nt][e] + bb;
    }
  }
}

// ---------------------------------------------------------------------------
extern "C" void kernel_launch(void* const* d_in, const int* in_sizes, int n_in,
                              void* d_out, int out_size, void* d_ws, size_t ws_size,
                              hipStream_t stream) {
  const float* x  = (const float*)d_in[0];
  const float* ck = (const float*)d_in[1];
  const float* cv = (const float*)d_in[2];
  const float* wp = (const float*)d_in[3];
  const float* bp = (const float*)d_in[4];
  float* out = (float*)d_out;

  char* ws = (char*)d_ws;
  _Float16* Kh  = (_Float16*)(ws);                            //  8 MiB: K * scale, f16
  _Float16* Vt  = (_Float16*)(ws + (((size_t)8)  << 20));     //  8 MiB: V transposed [b,h,d,s]
  _Float16* Wh  = (_Float16*)(ws + (((size_t)16) << 20));     //  2 MiB: w_proj f16
  _Float16* Att = (_Float16*)(ws + (((size_t)18) << 20));     //  8 MiB: attention out f16

  const int nK4 = (Bdim * Sdim * Wdim) / 4;   // 1,048,576
  const int nW4 = (Wdim * Wdim) / 4;          //   262,144
  cvt_f16_scaled<<<nK4 / 256, 256, 0, stream>>>(ck, Kh, QK_SCALE, nK4);
  cvt_f16_scaled<<<nW4 / 256, 256, 0, stream>>>(wp, Wh, 1.0f, nW4);
  transpose_v<<<dim3(Sdim / 64, Hn, Bdim), 256, 0, stream>>>(cv, Vt);
  flash_attn<<<dim3(Tdim / 128, Hn, Bdim), 256, 0, stream>>>(x, Kh, Vt, Att);
  proj_gemm<<<dim3((Bdim * Tdim) / 64, Wdim / 128), 256, 0, stream>>>(Att, Wh, bp, out);
}
